// TemporalGNN_22686017257800
// MI455X (gfx1250) — matrix operand order repaired and checked by reading, not compiled, not run on hardware
//
#include <hip/hip_runtime.h>
#include <cstdint>

// ---------------- problem constants ----------------
#define B_   4
#define S_   16
#define N_   1024
#define F_   64
#define H_   128
#define O_   64
#define NL_  3
#define NH_  4
#define CH_  128
#define CNUM (N_/CH_)                 // 8 chunks
#define RTOT (B_*S_*N_)               // 65536 total node-rows
#define BSN  (B_*S_)                  // 64
#define BN_SCALE 0.9999950000374996f  // 1/sqrt(1+1e-5)
#define QSCALE   0.17677669529663687f // 1/sqrt(32)

typedef __bf16 bf16;
typedef bf16  v16bf __attribute__((ext_vector_type(16)));
typedef bf16  v8bf  __attribute__((ext_vector_type(8)));
typedef float v8f   __attribute__((ext_vector_type(8)));

// ---------------- WMMA helpers (16x16x32 bf16 -> f32) ----------------
__device__ __forceinline__ v8f wmma_bf16(v16bf a, v16bf b, v8f c) {
  return __builtin_amdgcn_wmma_f32_16x16x32_bf16(false, a, false, b, (short)0, c, false, false);
}

// A fragment, row-major bf16 source [rows, ld], tile rows row0..row0+15, K in [k0,k0+32)
__device__ __forceinline__ v16bf load_a_bf16(const bf16* base, int ld, int row0, int k0, int lane) {
  int half = (lane >> 4) & 1, r = lane & 15;
  const bf16* p = base + (size_t)(row0 + r) * ld + k0 + half * 8;
  v8bf lo = *(const v8bf*)p;
  v8bf hi = *(const v8bf*)(p + 16);
  v16bf a;
#pragma unroll
  for (int t = 0; t < 8; ++t) { a[t] = lo[t]; a[t + 8] = hi[t]; }
  return a;
}

// A fragment from f32 source (convert on load)
__device__ __forceinline__ v16bf load_a_f32(const float* base, int ld, int row0, int k0, int lane) {
  int half = (lane >> 4) & 1, r = lane & 15;
  const float* p = base + (size_t)(row0 + r) * ld + k0 + half * 8;
  v8f lo = *(const v8f*)p;
  v8f hi = *(const v8f*)(p + 16);
  v16bf a;
#pragma unroll
  for (int t = 0; t < 8; ++t) { a[t] = (bf16)lo[t]; a[t + 8] = (bf16)hi[t]; }
  return a;
}

// B fragment: operand stored "transposed" as [ncols, K] row-major (ld = K stride),
// so for output column c the K-values are contiguous. elem t<8 -> K = k0 + 8*half + t, t>=8 -> K = k0 + 16 + 8*half + (t-8).
__device__ __forceinline__ v16bf load_b_bf16(const bf16* baseT, int ld, int col0, int k0, int lane) {
  int half = (lane >> 4) & 1, c = lane & 15;
  const bf16* p = baseT + (size_t)(col0 + c) * ld + k0 + half * 8;
  v8bf lo = *(const v8bf*)p;
  v8bf hi = *(const v8bf*)(p + 16);
  v16bf b;
#pragma unroll
  for (int t = 0; t < 8; ++t) { b[t] = lo[t]; b[t + 8] = hi[t]; }
  return b;
}

// ---------------- small prep kernels ----------------
__global__ __launch_bounds__(256) void prep_weights(
    const float* gcnW, const float* Wqkv, const float* Wo, const float* Wout,
    bf16* WgT, bf16* Wqkv_b, bf16* Wo_b, bf16* WoutT) {
  int i = blockIdx.x * 256 + threadIdx.x;
  if (i < NL_ * H_ * H_) {                       // WgT[l][n][k] = gcnW[l][k][n]
    int l = i / (H_ * H_), r = i % (H_ * H_), n = r / H_, k = r % H_;
    WgT[i] = (bf16)gcnW[((size_t)l * H_ + k) * H_ + n];
  }
  if (i < 3 * H_ * H_) Wqkv_b[i] = (bf16)Wqkv[i]; // used as [384 rows, K=H] directly
  if (i < H_ * H_)     Wo_b[i]   = (bf16)Wo[i];   // used as [H rows, K=H] directly
  if (i < O_ * H_) {                              // WoutT[n][k] = Wout[k][n]
    int n = i / H_, k = i % H_;
    WoutT[i] = (bf16)Wout[(size_t)k * O_ + n];
  }
}

// xv = (x*fi)@W_in + b_in ; layernorm over H -> hv [BS,H]
__global__ __launch_bounds__(128) void input_proj_ln(
    const float* x, const float* fi, const float* Win, const float* bin,
    const float* ln_g, const float* ln_b, float* hv) {
  int bs = blockIdx.x, j = threadIdx.x;
  const float* xr = x + (size_t)bs * F_;
  float acc = bin[j];
  for (int k = 0; k < F_; ++k) acc += xr[k] * fi[k] * Win[(size_t)k * H_ + j];
  __shared__ float s1[H_];
  s1[j] = acc; __syncthreads();
  for (int m = 64; m > 0; m >>= 1) { if (j < m) s1[j] += s1[j + m]; __syncthreads(); }
  float mu = s1[0] / H_; __syncthreads();
  float d = acc - mu;
  s1[j] = d * d; __syncthreads();
  for (int m = 64; m > 0; m >>= 1) { if (j < m) s1[j] += s1[j + m]; __syncthreads(); }
  float var = s1[0] / H_;
  hv[(size_t)bs * H_ + j] = d * rsqrtf(var + 1e-5f) * ln_g[j] + ln_b[j];
}

__global__ __launch_bounds__(256) void bcast_h(const float* hv, float* h) {
  size_t idx = (size_t)blockIdx.x * 256 + threadIdx.x;
  if (idx >= (size_t)RTOT * H_) return;
  size_t bs = idx / ((size_t)N_ * H_);
  int j = (int)(idx % H_);
  h[idx] = hv[bs * H_ + j];
}

__global__ __launch_bounds__(256) void skip_init(const float* hv, float* h) {
  size_t idx = (size_t)blockIdx.x * 256 + threadIdx.x;
  if (idx >= (size_t)RTOT * H_) return;
  size_t bs = idx / ((size_t)N_ * H_);
  int j = (int)(idx % H_);
  h[idx] += hv[bs * H_ + j];
}

// dinv[row] = rsqrt(1 + sum_j adj[row][j])  (one wave32 per row)
__global__ __launch_bounds__(256) void degree_kernel(const float* adj, float* dinv) {
  int row  = blockIdx.x * 8 + (threadIdx.x >> 5);
  int lane = threadIdx.x & 31;
  const float* r = adj + (size_t)row * N_;
  float s = 0.f;
  for (int j = lane; j < N_; j += 32) s += r[j];
#pragma unroll
  for (int m = 16; m > 0; m >>= 1) s += __shfl_xor(s, m, 32);
  if (lane == 0) dinv[row] = rsqrtf(s + 1.0f);
}

// ---------------- GEMM kernels ----------------
// One wave owns a 16-row stripe and ALL column tiles of that stripe:
// the A fragment is built once per K-step and reused for every column tile,
// so the streaming operand is read exactly once.

// hwT[bs][j][n] = (h @ WgT_l)[bs,n,j], stored transposed bf16 so SpMM B-frags are contiguous
__global__ __launch_bounds__(256) void gemm_hwT(const float* h, const bf16* WgT_l, bf16* hwT) {
  int rt = blockIdx.x * 8 + (threadIdx.x >> 5);    // 4096 row tiles
  int lane = threadIdx.x & 31, half = lane >> 4, l16 = lane & 15;
  v8f acc[8] = {};
  for (int kb = 0; kb < H_; kb += 32) {
    v16bf a = load_a_f32(h, H_, rt * 16, kb, lane);
#pragma unroll
    for (int u = 0; u < 8; ++u) {
      v16bf b = load_b_bf16(WgT_l, H_, u * 16, kb, lane);
      acc[u] = wmma_bf16(a, b, acc[u]);
    }
  }
  int gr0 = rt * 16 + half * 8;
  int bs = gr0 >> 10, n0 = gr0 & (N_ - 1);
#pragma unroll
  for (int u = 0; u < 8; ++u) {
    int j = u * 16 + l16;
    v8bf pack;
#pragma unroll
    for (int v = 0; v < 8; ++v) pack[v] = (bf16)acc[u][v];
    *(v8bf*)(hwT + ((size_t)bs * H_ + j) * N_ + n0) = pack;  // contiguous in n
  }
}

// hc1 = BN( A_norm @ hw + gcn_b ), A_norm built on the fly from adj + dinv (bf16 WMMA).
// adj is the only HBM-resident stream: read exactly once per layer.
__global__ __launch_bounds__(256) void spmm_gcn(
    const float* adj, const float* dinv, const bf16* hwT,
    const float* gcn_b_l, const float* bn_g_l, const float* bn_b_l, bf16* hc1) {
  int bs = blockIdx.y;
  __shared__ float sdinv[N_];
  for (int t = threadIdx.x; t < N_; t += 256) sdinv[t] = dinv[(size_t)bs * N_ + t];
  __syncthreads();
  int rt = blockIdx.x * 8 + (threadIdx.x >> 5);    // 64 row tiles per (b,s)
  int lane = threadIdx.x & 31, half = lane >> 4, l16 = lane & 15;
  int i = rt * 16 + l16;
  const float* arow = adj + ((size_t)bs * N_ + i) * N_;
  float di = sdinv[i];
  const bf16* hwTb = hwT + (size_t)bs * H_ * N_;
  v8f acc[8] = {};
  for (int kb = 0; kb < N_; kb += 32) {
    int k0 = kb + half * 8;
    v8f lo = *(const v8f*)(arow + k0);
    v8f hi = *(const v8f*)(arow + k0 + 16);
    __builtin_prefetch((const void*)(arow + kb + 512), 0, 3);  // ~2KB ahead per lane-row
    v16bf a;
#pragma unroll
    for (int t = 0; t < 8; ++t) {
      int kc = k0 + t;
      a[t]     = (bf16)((lo[t] + (kc == i ? 1.0f : 0.0f)) * di * sdinv[kc]);
      int kc2 = k0 + 16 + t;
      a[t + 8] = (bf16)((hi[t] + (kc2 == i ? 1.0f : 0.0f)) * di * sdinv[kc2]);
    }
#pragma unroll
    for (int u = 0; u < 8; ++u) {
      v16bf b = load_b_bf16(hwTb, N_, u * 16, kb, lane);
      acc[u] = wmma_bf16(a, b, acc[u]);
    }
  }
  bf16* outr = hc1 + ((size_t)bs * N_ + rt * 16 + half * 8) * H_;
#pragma unroll
  for (int u = 0; u < 8; ++u) {
    int j = u * 16 + l16;
    float gb = gcn_b_l[j];
    float g  = bn_g_l[j] * BN_SCALE;
    float bb = bn_b_l[j];
#pragma unroll
    for (int v = 0; v < 8; ++v) outr[(size_t)v * H_ + j] = (bf16)((acc[u][v] + gb) * g + bb);
  }
}

// qkv = hc1 @ Wqkv^T + bqkv ; each wave takes one 128-col group (cg=0:q, 1:k, 2:v)
// so the split branch is wave-uniform. v stored [bs,chunk,head,dh,token].
__global__ __launch_bounds__(256) void gemm_qkv(
    const bf16* hc1, const bf16* Wqkv_b, const float* bqkv,
    bf16* qb, bf16* kbuf, bf16* vT) {
  int w = blockIdx.x * 8 + (threadIdx.x >> 5);     // 12288 waves
  int lane = threadIdx.x & 31, half = lane >> 4, l16 = lane & 15;
  int cg = w % 3, rt = w / 3;                      // col group of 8 tiles, 4096 row tiles
  v8f acc[8] = {};
  for (int kb = 0; kb < H_; kb += 32) {
    v16bf a = load_a_bf16(hc1, H_, rt * 16, kb, lane);
#pragma unroll
    for (int u = 0; u < 8; ++u) {
      v16bf b = load_b_bf16(Wqkv_b, H_, cg * 128 + u * 16, kb, lane);
      acc[u] = wmma_bf16(a, b, acc[u]);
    }
  }
  int gr0 = rt * 16 + half * 8;
  int bs = gr0 >> 10, n0 = gr0 & (N_ - 1);
  if (cg == 0) {
    bf16* o = qb + (size_t)gr0 * H_;
#pragma unroll
    for (int u = 0; u < 8; ++u) {
      int j = u * 16 + l16;
      float bias = bqkv[j];
#pragma unroll
      for (int v = 0; v < 8; ++v) o[(size_t)v * H_ + j] = (bf16)((acc[u][v] + bias) * QSCALE);
    }
  } else if (cg == 1) {
    bf16* o = kbuf + (size_t)gr0 * H_;
#pragma unroll
    for (int u = 0; u < 8; ++u) {
      int j = u * 16 + l16;
      float bias = bqkv[128 + j];
#pragma unroll
      for (int v = 0; v < 8; ++v) o[(size_t)v * H_ + j] = (bf16)(acc[u][v] + bias);
    }
  } else {
    int chunk = n0 >> 7, m0 = n0 & (CH_ - 1);
#pragma unroll
    for (int u = 0; u < 8; ++u) {
      int jj = u * 16 + l16;                       // 0..127 within v block
      int head = jj >> 5, d = jj & 31;
      float bias = bqkv[256 + jj];
      v8bf pack;
#pragma unroll
      for (int v = 0; v < 8; ++v) pack[v] = (bf16)(acc[u][v] + bias);
      *(v8bf*)(vT + ((((size_t)bs * CNUM + chunk) * NH_ + head) * 32 + d) * CH_ + m0) = pack;
    }
  }
}

// per (b,s,chunk,head): scores = q@k^T (WMMA), softmax in LDS, o = attn@v (WMMA)
__global__ __launch_bounds__(256) void attention_kernel(
    const bf16* qb, const bf16* kbuf, const bf16* vT, bf16* att) {
  extern __shared__ char smem_raw[];
  float* sc = (float*)smem_raw;                          // 128*128 f32 = 64KB
  bf16*  at = (bf16*)(smem_raw + (size_t)CH_ * CH_ * 4); // 128*128 bf16 = 32KB
  int id = blockIdx.x;
  int head = id % NH_; id /= NH_;
  int chunk = id % CNUM; int bs = id / CNUM;
  int tid = threadIdx.x, wave = tid >> 5, lane = tid & 31;
  int half = lane >> 4, l16 = lane & 15;
  size_t rowbase = ((size_t)bs * N_ + chunk * CH_) * H_ + head * 32;
  const bf16* qbase = qb   + rowbase;   // [128 tok, K=32] ld=H_
  const bf16* kbase = kbuf + rowbase;   // B operand: row m contiguous in d
  // ---- scores (K=32 -> single WMMA per tile) ----
  v16bf aq = load_a_bf16(qbase, H_, wave * 16, 0, lane);
#pragma unroll
  for (int ctt = 0; ctt < 8; ++ctt) {
    v16bf bk = load_b_bf16(kbase, H_, ctt * 16, 0, lane);
    v8f c = {};
    c = wmma_bf16(aq, bk, c);
#pragma unroll
    for (int v = 0; v < 8; ++v)
      sc[(size_t)(wave * 16 + half * 8 + v) * CH_ + ctt * 16 + l16] = c[v];
  }
  __syncthreads();
  // ---- softmax, row per thread (threads 0..127) ----
  if (tid < CH_) {
    float* row = sc + (size_t)tid * CH_;
    float mx = -1e30f;
    for (int jj = 0; jj < CH_; ++jj) mx = fmaxf(mx, row[jj]);
    float s = 0.f;
    for (int jj = 0; jj < CH_; ++jj) { float e = __expf(row[jj] - mx); row[jj] = e; s += e; }
    float inv = 1.0f / s;
    for (int jj = 0; jj < CH_; ++jj) at[(size_t)tid * CH_ + jj] = (bf16)(row[jj] * inv);
  }
  __syncthreads();
  // ---- o = attn @ v  (out [128,32], 2 col-tiles per wave) ----
  const bf16* vbase = vT + (((size_t)bs * CNUM + chunk) * NH_ + head) * 32 * CH_; // [32,128]
#pragma unroll
  for (int ctt = 0; ctt < 2; ++ctt) {
    v8f c = {};
    for (int kb = 0; kb < CH_; kb += 32) {
      v16bf aa = load_a_bf16(at, CH_, wave * 16, kb, lane);
      v16bf bv = load_b_bf16(vbase, CH_, ctt * 16, kb, lane);
      c = wmma_bf16(aa, bv, c);
    }
    int j = head * 32 + ctt * 16 + l16;
#pragma unroll
    for (int v = 0; v < 8; ++v) {
      int m = wave * 16 + half * 8 + v;
      att[((size_t)bs * N_ + chunk * CH_ + m) * H_ + j] = (bf16)c[v];
    }
  }
}

// h += relu( att @ Wo^T + bo )   (skip connection)
__global__ __launch_bounds__(256) void gemm_oproj(
    const bf16* att, const bf16* Wo_b, const float* bo, float* h) {
  int rt = blockIdx.x * 8 + (threadIdx.x >> 5);
  int lane = threadIdx.x & 31, half = lane >> 4, l16 = lane & 15;
  v8f acc[8] = {};
  for (int kb = 0; kb < H_; kb += 32) {
    v16bf a = load_a_bf16(att, H_, rt * 16, kb, lane);
#pragma unroll
    for (int u = 0; u < 8; ++u) {
      v16bf b = load_b_bf16(Wo_b, H_, u * 16, kb, lane);
      acc[u] = wmma_bf16(a, b, acc[u]);
    }
  }
  float* orow = h + (size_t)(rt * 16 + half * 8) * H_;
#pragma unroll
  for (int u = 0; u < 8; ++u) {
    int j = u * 16 + l16;
    float bias = bo[j];
#pragma unroll
    for (int v = 0; v < 8; ++v) {
      float val = acc[u][v] + bias;
      val = val > 0.f ? val : 0.f;
      orow[(size_t)v * H_ + j] += val;
    }
  }
}

// node = (h @ W_out + b_out) * mask[bs]
__global__ __launch_bounds__(256) void gemm_out(
    const float* h, const bf16* WoutT, const float* b_out, const float* mask, float* node) {
  int rt = blockIdx.x * 8 + (threadIdx.x >> 5);
  int lane = threadIdx.x & 31, half = lane >> 4, l16 = lane & 15;
  v8f acc[4] = {};
  for (int kb = 0; kb < H_; kb += 32) {
    v16bf a = load_a_f32(h, H_, rt * 16, kb, lane);
#pragma unroll
    for (int u = 0; u < 4; ++u) {
      v16bf b = load_b_bf16(WoutT, H_, u * 16, kb, lane);
      acc[u] = wmma_bf16(a, b, acc[u]);
    }
  }
  int gr0 = rt * 16 + half * 8;
  float m = mask[gr0 >> 10];
  float* orow = node + (size_t)gr0 * O_;
#pragma unroll
  for (int u = 0; u < 4; ++u) {
    int j = u * 16 + l16;
    float bias = b_out[j];
#pragma unroll
    for (int v = 0; v < 8; ++v) orow[(size_t)v * O_ + j] = (acc[u][v] + bias) * m;
  }
}

// graph[bs][o] = mean_n node[bs][n][o]  (node already masked)
__global__ __launch_bounds__(256) void graph_mean(const float* node, float* graph) {
  int bs = blockIdx.x, tid = threadIdx.x;
  int o = tid & 63, seg = tid >> 6;
  float s = 0.f;
  for (int n = seg * 256; n < (seg + 1) * 256; ++n)
    s += node[((size_t)bs * N_ + n) * O_ + o];
  __shared__ float red[256];
  red[tid] = s; __syncthreads();
  if (seg == 0)
    graph[(size_t)bs * O_ + o] = (red[o] + red[o + 64] + red[o + 128] + red[o + 192]) * (1.0f / N_);
}

// ---------------- host launcher ----------------
extern "C" void kernel_launch(void* const* d_in, const int* in_sizes, int n_in,
                              void* d_out, int out_size, void* d_ws, size_t ws_size,
                              hipStream_t stream) {
  const float* x     = (const float*)d_in[0];
  const float* adj   = (const float*)d_in[1];
  const float* mask  = (const float*)d_in[2];
  const float* fi    = (const float*)d_in[3];
  const float* W_in  = (const float*)d_in[4];
  const float* b_in  = (const float*)d_in[5];
  const float* ln_g  = (const float*)d_in[6];
  const float* ln_b  = (const float*)d_in[7];
  const float* gcn_W = (const float*)d_in[8];
  const float* gcn_b = (const float*)d_in[9];
  const float* bn_g  = (const float*)d_in[10];
  const float* bn_b  = (const float*)d_in[11];
  const float* Wqkv  = (const float*)d_in[12];
  const float* bqkv  = (const float*)d_in[13];
  const float* Wo    = (const float*)d_in[14];
  const float* bo    = (const float*)d_in[15];
  const float* W_out = (const float*)d_in[16];
  const float* b_out = (const float*)d_in[17];

  float* node  = (float*)d_out;                         // [B,S,N,O]
  float* graph = node + (size_t)B_ * S_ * N_ * O_;      // [B,S,O]

  char* ws = (char*)d_ws;
  size_t off = 0;
  auto alloc = [&](size_t bytes) -> char* {
    char* p = ws + off; off = (off + bytes + 255) & ~(size_t)255; return p;
  };
  float* dinv   = (float*)alloc((size_t)RTOT * 4);
  float* hv     = (float*)alloc((size_t)BSN * H_ * 4);
  float* h      = (float*)alloc((size_t)RTOT * H_ * 4);          // 32 MB
  bf16*  hwT    = (bf16*) alloc((size_t)BSN * H_ * N_ * 2);      // 16 MB
  bf16*  hc1    = (bf16*) alloc((size_t)RTOT * H_ * 2);
  bf16*  qb     = (bf16*) alloc((size_t)RTOT * H_ * 2);
  bf16*  kbuf   = (bf16*) alloc((size_t)RTOT * H_ * 2);
  bf16*  vT     = (bf16*) alloc((size_t)RTOT * H_ * 2);
  bf16*  att    = (bf16*) alloc((size_t)RTOT * H_ * 2);
  bf16*  WgT    = (bf16*) alloc((size_t)NL_ * H_ * H_ * 2);
  bf16*  Wqkv_b = (bf16*) alloc((size_t)3 * H_ * H_ * 2);
  bf16*  Wo_b   = (bf16*) alloc((size_t)H_ * H_ * 2);
  bf16*  WoutT  = (bf16*) alloc((size_t)O_ * H_ * 2);

  prep_weights<<<192, 256, 0, stream>>>(gcn_W, Wqkv, Wo, W_out, WgT, Wqkv_b, Wo_b, WoutT);
  input_proj_ln<<<BSN, H_, 0, stream>>>(x, fi, W_in, b_in, ln_g, ln_b, hv);
  bcast_h<<<(RTOT * H_) / 256, 256, 0, stream>>>(hv, h);
  degree_kernel<<<RTOT / 8, 256, 0, stream>>>(adj, dinv);

  for (int l = 0; l < NL_; ++l) {
    gemm_hwT<<<512, 256, 0, stream>>>(h, WgT + (size_t)l * H_ * H_, hwT);
    spmm_gcn<<<dim3(8, BSN), 256, 0, stream>>>(adj, dinv, hwT,
        gcn_b + (size_t)l * H_, bn_g + (size_t)l * H_, bn_b + (size_t)l * H_, hc1);
    gemm_qkv<<<1536, 256, 0, stream>>>(hc1, Wqkv_b, bqkv, qb, kbuf, vT);
    attention_kernel<<<BSN * CNUM * NH_, 256, (size_t)CH_ * CH_ * 6, stream>>>(qb, kbuf, vT, att);
    gemm_oproj<<<512, 256, 0, stream>>>(att, Wo_b, bo, h);
  }

  skip_init<<<(RTOT * H_) / 256, 256, 0, stream>>>(hv, h);
  gemm_out<<<512, 256, 0, stream>>>(h, WoutT, b_out, mask, node);
  graph_mean<<<BSN, 256, 0, stream>>>(node, graph);
}
